// LSTMLayer_1391569404088
// MI455X (gfx1250) — compile-verified, matrix operand order checked by reading, not run on hardware
//
#include <hip/hip_runtime.h>
#include <hip/hip_bf16.h>

// ---------------------------------------------------------------------------
// Problem constants (from reference)
// ---------------------------------------------------------------------------
#define BB 256   // batch
#define TT 128   // timesteps
#define II 512   // input features
#define HH 1024  // hidden
#define CC 2048  // classes

typedef __attribute__((ext_vector_type(16))) __bf16 v16bf;
typedef __attribute__((ext_vector_type(8)))  __bf16 v8bf;
typedef __attribute__((ext_vector_type(8)))  float  v8f;
typedef __attribute__((ext_vector_type(4)))  int    i4;

// fp32 -> bf16 with round-to-nearest-even (used only in pre-pass / epilogues)
__device__ __forceinline__ __bf16 f2bf(float f) {
    unsigned u = __builtin_bit_cast(unsigned, f);
    u += 0x7fffu + ((u >> 16) & 1u);
    unsigned short s = (unsigned short)(u >> 16);
    return __builtin_bit_cast(__bf16, s);
}

// Load one 16-element bf16 fragment for the 16-bit A/B WMMA layout
// (ISA 7.12.2): lane group "hig" selects K offsets {hig*8..+7} and
// {16+hig*8..+7} within a K=32 slab.  Two 16B loads, no conversion VALU.
struct BFragBits { i4 a, b; };
__device__ __forceinline__ v16bf load_frag(const __bf16* __restrict__ rowk, int hig) {
    BFragBits f;
    f.a = *(const i4*)(rowk + hig * 8);
    f.b = *(const i4*)(rowk + 16 + hig * 8);
    return __builtin_bit_cast(v16bf, f);
}

__device__ __forceinline__ v8f wmma_bf16(v16bf a, v16bf b, v8f c) {
    // (neg_a, A, neg_b, B, c_mod, C, reuse_a, reuse_b)
    return __builtin_amdgcn_wmma_f32_16x16x32_bf16(false, a, false, b, (short)0, c,
                                                   false, false);
}

__device__ __forceinline__ float sigm(float x) {
    return 1.0f / (1.0f + __expf(-x));
}
__device__ __forceinline__ float fast_tanh(float x) {
    return 1.0f - 2.0f / (__expf(2.0f * x) + 1.0f);
}

// ---------------------------------------------------------------------------
// Pre-pass: fp32 -> bf16 buffer conversion, 8 elements / thread, b128 stores.
// ---------------------------------------------------------------------------
__global__ void __launch_bounds__(256)
f32_to_bf16_kernel(const float* __restrict__ src, __bf16* __restrict__ dst, int n8) {
    int i = blockIdx.x * 256 + threadIdx.x;
    if (i >= n8) return;
    const float4* s = (const float4*)src + (size_t)2 * i;
    float4 q0 = s[0], q1 = s[1];
    v8bf o;
    o[0] = f2bf(q0.x); o[1] = f2bf(q0.y); o[2] = f2bf(q0.z); o[3] = f2bf(q0.w);
    o[4] = f2bf(q1.x); o[5] = f2bf(q1.y); o[6] = f2bf(q1.z); o[7] = f2bf(q1.w);
    *((v8bf*)dst + i) = o;
}

// ---------------------------------------------------------------------------
// Fused bidirectional LSTM step: z = x_t@W_ih^T + h_{t-1}@W_hh^T + b, gates,
// c/h update.  One wave = 32(batch) x 16(H-col) tile: 2 M-subtiles sharing
// every weight fragment, 4 gate accumulators each (8 WMMA / K-slab).
// grid = (HH/16, 1, 2 dirs), block = 256 (8 waves cover all 256 batch rows).
// ---------------------------------------------------------------------------
__global__ void __launch_bounds__(256)
lstm_step_kernel(const __bf16* __restrict__ x,
                 const __bf16* __restrict__ w_ih_f, const __bf16* __restrict__ w_hh_f,
                 const float* __restrict__ b_ih_f, const float* __restrict__ b_hh_f,
                 const __bf16* __restrict__ w_ih_b, const __bf16* __restrict__ w_hh_b,
                 const float* __restrict__ b_ih_b, const float* __restrict__ b_hh_b,
                 __bf16* __restrict__ hs_f, __bf16* __restrict__ hs_b,
                 float* __restrict__ c_f, float* __restrict__ c_b,
                 int s) {
    const int wave = threadIdx.x >> 5;
    const int lane = threadIdx.x & 31;
    const int hig  = lane >> 4;
    const int lo   = lane & 15;

    const int tileN  = blockIdx.x;      // 0..HH/16-1
    const int tileM0 = wave * 2;        // 0,2,..,14
    const int dir    = blockIdx.z;

    const __bf16* w_ih = dir ? w_ih_b : w_ih_f;
    const __bf16* w_hh = dir ? w_hh_b : w_hh_f;
    const float*  b_ih = dir ? b_ih_b : b_ih_f;
    const float*  b_hh = dir ? b_hh_b : b_hh_f;
    __bf16* hs   = dir ? hs_b : hs_f;
    float*  cbuf = dir ? c_b : c_f;

    const int  t      = dir ? (TT - 1 - s) : s;
    const int  t_prev = dir ? (t + 1) : (t - 1);
    const bool first  = (s == 0);

    const int n  = tileN * 16 + lo;         // weight row / output column in H
    const int m0 = tileM0 * 16 + lo;        // batch rows for the two A frags
    const int m1 = m0 + 16;

    v8f acc[2][4];
#pragma unroll
    for (int g = 0; g < 4; ++g) {
        float bg = b_ih[g * HH + n] + b_hh[g * HH + n];
#pragma unroll
        for (int v = 0; v < 8; ++v) { acc[0][g][v] = bg; acc[1][g][v] = bg; }
    }

    // ---- input projection: x[b][t][:] @ W_ih^T  (K = II) ----
    {
        const __bf16* arow0 = x + ((size_t)m0 * TT + t) * II;
        const __bf16* arow1 = x + ((size_t)m1 * TT + t) * II;
#pragma unroll 2
        for (int kt = 0; kt < II; kt += 32) {
            v16bf a0 = load_frag(arow0 + kt, hig);
            v16bf a1 = load_frag(arow1 + kt, hig);
            __builtin_prefetch(arow0 + kt + 32, 0, 0);
#pragma unroll
            for (int g = 0; g < 4; ++g) {
                v16bf b = load_frag(w_ih + (size_t)(g * HH + n) * II + kt, hig);
                acc[0][g] = wmma_bf16(a0, b, acc[0][g]);
                acc[1][g] = wmma_bf16(a1, b, acc[1][g]);
            }
        }
    }

    // ---- hidden projection: h_{t-1} @ W_hh^T  (K = HH) ----
    if (!first) {
        const __bf16* hrow0 = hs + ((size_t)t_prev * BB + m0) * HH;
        const __bf16* hrow1 = hs + ((size_t)t_prev * BB + m1) * HH;
#pragma unroll 2
        for (int kt = 0; kt < HH; kt += 32) {
            v16bf a0 = load_frag(hrow0 + kt, hig);
            v16bf a1 = load_frag(hrow1 + kt, hig);
            __builtin_prefetch(hrow0 + kt + 32, 0, 0);
#pragma unroll
            for (int g = 0; g < 4; ++g) {
                v16bf b = load_frag(w_hh + (size_t)(g * HH + n) * HH + kt, hig);
                acc[0][g] = wmma_bf16(a0, b, acc[0][g]);
                acc[1][g] = wmma_bf16(a1, b, acc[1][g]);
            }
        }
    }

    // ---- gate nonlinearity + state update (C/D layout: row = v + 8*hig) ----
    const int col = tileN * 16 + lo;
#pragma unroll
    for (int u = 0; u < 2; ++u) {
#pragma unroll
        for (int v = 0; v < 8; ++v) {
            const int mb = (tileM0 + u) * 16 + hig * 8 + v;
            float zi = acc[u][0][v], zf = acc[u][1][v];
            float zg = acc[u][2][v], zo = acc[u][3][v];
            float cprev = first ? 0.0f : cbuf[(size_t)mb * HH + col];
            float cn = sigm(zf) * cprev + sigm(zi) * fast_tanh(zg);
            float hn = sigm(zo) * fast_tanh(cn);
            cbuf[(size_t)mb * HH + col] = cn;
            hs[((size_t)t * BB + mb) * HH + col] = f2bf(hn);
        }
    }
}

// ---------------------------------------------------------------------------
// Attention: att[t,b] = sigmoid(3 * (dot(w_att, [h_f;h_b]) + b_att)).
// One wave per (t,b); vectorized bf16 loads; wave-shuffle reduction.
// grid = T*B/8, block = 256.
// ---------------------------------------------------------------------------
__global__ void __launch_bounds__(256)
attention_kernel(const __bf16* __restrict__ hs_f, const __bf16* __restrict__ hs_b,
                 const float* __restrict__ w_att, const float* __restrict__ b_att,
                 float* __restrict__ att) {
    const int wave = threadIdx.x >> 5;
    const int lane = threadIdx.x & 31;
    const int p    = blockIdx.x * 8 + wave;        // p = t*BB + b
    const __bf16* hf = hs_f + (size_t)p * HH;
    const __bf16* hb = hs_b + (size_t)p * HH;
    float s = 0.0f;
#pragma unroll
    for (int base = 0; base < HH; base += 256) {
        const int k = base + lane * 8;
        v8bf vf = *(const v8bf*)(hf + k);
        v8bf vb = *(const v8bf*)(hb + k);
#pragma unroll
        for (int j = 0; j < 8; ++j) {
            s += (float)vf[j] * w_att[k + j];
            s += (float)vb[j] * w_att[HH + k + j];
        }
    }
#pragma unroll
    for (int off = 16; off > 0; off >>= 1)
        s += __shfl_xor(s, off, 32);
    if (lane == 0)
        att[p] = 1.0f / (1.0f + __expf(-3.0f * (s + b_att[0])));
}

// ---------------------------------------------------------------------------
// Attention-gated custom cell step:
// h' = a*relu(x_t@Wi2h^T + b_i2h + h@Wh2h^T + b_h2h) + (1-a)*h
// grid = (HH/16, BB/128), block = 256.
// ---------------------------------------------------------------------------
__global__ void __launch_bounds__(256)
cell_step_kernel(const __bf16* __restrict__ x,
                 const __bf16* __restrict__ w_i2h, const float* __restrict__ b_i2h,
                 const __bf16* __restrict__ w_h2h, const float* __restrict__ b_h2h,
                 const float* __restrict__ att,
                 const __bf16* __restrict__ h_prev, __bf16* __restrict__ h_next,
                 int t) {
    const int wave = threadIdx.x >> 5;
    const int lane = threadIdx.x & 31;
    const int hig  = lane >> 4;
    const int lo   = lane & 15;
    const int tileN = blockIdx.x;
    const int tileM = blockIdx.y * 8 + wave;
    const bool first = (t == 0);

    const int n = tileN * 16 + lo;
    const int m = tileM * 16 + lo;

    v8f acc;
    {
        float bg = b_i2h[n] + b_h2h[n];
#pragma unroll
        for (int v = 0; v < 8; ++v) acc[v] = bg;
    }

    const __bf16* arow = x + ((size_t)m * TT + t) * II;
#pragma unroll 2
    for (int kt = 0; kt < II; kt += 32) {
        v16bf a = load_frag(arow + kt, hig);
        v16bf b = load_frag(w_i2h + (size_t)n * II + kt, hig);
        acc = wmma_bf16(a, b, acc);
    }
    if (!first) {
        const __bf16* hrow = h_prev + (size_t)m * HH;
#pragma unroll 2
        for (int kt = 0; kt < HH; kt += 32) {
            v16bf a = load_frag(hrow + kt, hig);
            v16bf b = load_frag(w_h2h + (size_t)n * HH + kt, hig);
            acc = wmma_bf16(a, b, acc);
        }
    }

    const int col = tileN * 16 + lo;
#pragma unroll
    for (int v = 0; v < 8; ++v) {
        const int mb = tileM * 16 + hig * 8 + v;
        float r = acc[v];
        r = r > 0.0f ? r : 0.0f;                       // relu
        float a = att[(size_t)t * BB + mb];
        float hp = first ? 0.0f : (float)h_prev[(size_t)mb * HH + col];
        h_next[(size_t)mb * HH + col] = f2bf(a * r + (1.0f - a) * hp);
    }
}

// ---------------------------------------------------------------------------
// Final FC: out = h_last @ W_fc^T + b_fc.  Wave tile = 16 x 64 (4 N-subtiles).
// grid = (CC/64, BB/128), block = 256.
// ---------------------------------------------------------------------------
__global__ void __launch_bounds__(256)
fc_kernel(const __bf16* __restrict__ A, const __bf16* __restrict__ W,
          const float* __restrict__ bias, float* __restrict__ out) {
    const int wave = threadIdx.x >> 5;
    const int lane = threadIdx.x & 31;
    const int hig  = lane >> 4;
    const int lo   = lane & 15;
    const int tileM  = blockIdx.y * 8 + wave;
    const int tileN0 = blockIdx.x * 4;
    const int m = tileM * 16 + lo;

    v8f acc[4];
#pragma unroll
    for (int j = 0; j < 4; ++j) {
        float bg = bias[(tileN0 + j) * 16 + lo];
#pragma unroll
        for (int v = 0; v < 8; ++v) acc[j][v] = bg;
    }

    const __bf16* arow = A + (size_t)m * HH;
#pragma unroll 2
    for (int kt = 0; kt < HH; kt += 32) {
        v16bf a = load_frag(arow + kt, hig);
#pragma unroll
        for (int j = 0; j < 4; ++j) {
            const int nn = (tileN0 + j) * 16 + lo;
            v16bf b = load_frag(W + (size_t)nn * HH + kt, hig);
            acc[j] = wmma_bf16(a, b, acc[j]);
        }
    }

#pragma unroll
    for (int j = 0; j < 4; ++j) {
        const int col = (tileN0 + j) * 16 + lo;
#pragma unroll
        for (int v = 0; v < 8; ++v) {
            const int mb = tileM * 16 + hig * 8 + v;
            out[(size_t)mb * CC + col] = acc[j][v];
        }
    }
}

// ---------------------------------------------------------------------------
// Host-side launch
// ---------------------------------------------------------------------------
extern "C" void kernel_launch(void* const* d_in, const int* in_sizes, int n_in,
                              void* d_out, int out_size, void* d_ws, size_t ws_size,
                              hipStream_t stream) {
    const float* x      = (const float*)d_in[0];
    const float* w_ih_f = (const float*)d_in[1];
    const float* w_hh_f = (const float*)d_in[2];
    const float* b_ih_f = (const float*)d_in[3];
    const float* b_hh_f = (const float*)d_in[4];
    const float* w_ih_b = (const float*)d_in[5];
    const float* w_hh_b = (const float*)d_in[6];
    const float* b_ih_b = (const float*)d_in[7];
    const float* b_hh_b = (const float*)d_in[8];
    const float* w_att  = (const float*)d_in[9];
    const float* b_att  = (const float*)d_in[10];
    const float* w_i2h  = (const float*)d_in[11];
    const float* b_i2h  = (const float*)d_in[12];
    const float* w_h2h  = (const float*)d_in[13];
    const float* b_h2h  = (const float*)d_in[14];
    const float* w_fc   = (const float*)d_in[15];
    const float* b_fc   = (const float*)d_in[16];

    // ---- workspace carve: fp32 region first, then bf16 region ----
    float* wsf  = (float*)d_ws;
    float* c_f  = wsf;                                 // B*H
    float* c_b  = c_f + (size_t)BB * HH;               // B*H
    float* att  = c_b + (size_t)BB * HH;               // T*B
    __bf16* wsb = (__bf16*)(att + (size_t)TT * BB);

    __bf16* x_bf    = wsb;                                   // B*T*I
    __bf16* wihf_bf = x_bf    + (size_t)BB * TT * II;        // 4H*I
    __bf16* whhf_bf = wihf_bf + (size_t)4 * HH * II;         // 4H*H
    __bf16* wihb_bf = whhf_bf + (size_t)4 * HH * HH;         // 4H*I
    __bf16* whhb_bf = wihb_bf + (size_t)4 * HH * II;         // 4H*H
    __bf16* wi2h_bf = whhb_bf + (size_t)4 * HH * HH;         // H*I
    __bf16* wh2h_bf = wi2h_bf + (size_t)HH * II;             // H*H
    __bf16* wfc_bf  = wh2h_bf + (size_t)HH * HH;             // C*H
    __bf16* hs_f    = wfc_bf  + (size_t)CC * HH;             // T*B*H
    __bf16* hs_b    = hs_f    + (size_t)TT * BB * HH;        // T*B*H
    __bf16* hA      = hs_b    + (size_t)TT * BB * HH;        // B*H
    __bf16* hB      = hA      + (size_t)BB * HH;             // B*H

    const dim3 blk(256);

    // ---- one-time bf16 conversion of all GEMM operands ----
    auto conv = [&](const float* s, __bf16* d, size_t n) {
        int n8 = (int)(n / 8);
        f32_to_bf16_kernel<<<dim3((n8 + 255) / 256), blk, 0, stream>>>(s, d, n8);
    };
    conv(x,      x_bf,    (size_t)BB * TT * II);
    conv(w_ih_f, wihf_bf, (size_t)4 * HH * II);
    conv(w_hh_f, whhf_bf, (size_t)4 * HH * HH);
    conv(w_ih_b, wihb_bf, (size_t)4 * HH * II);
    conv(w_hh_b, whhb_bf, (size_t)4 * HH * HH);
    conv(w_i2h,  wi2h_bf, (size_t)HH * II);
    conv(w_h2h,  wh2h_bf, (size_t)HH * HH);
    conv(w_fc,   wfc_bf,  (size_t)CC * HH);

    // ---- bidirectional LSTM, one fused launch per timestep ----
    for (int s = 0; s < TT; ++s) {
        lstm_step_kernel<<<dim3(HH / 16, 1, 2), blk, 0, stream>>>(
            x_bf, wihf_bf, whhf_bf, b_ih_f, b_hh_f,
            wihb_bf, whhb_bf, b_ih_b, b_hh_b,
            hs_f, hs_b, c_f, c_b, s);
    }

    // ---- attention weights over the whole sequence ----
    attention_kernel<<<dim3((TT * BB) / 8), blk, 0, stream>>>(
        hs_f, hs_b, w_att, b_att, att);

    // ---- attention-gated custom cell (ping-pong h buffers) ----
    for (int s = 0; s < TT; ++s) {
        const __bf16* hp = (s & 1) ? hB : hA;
        __bf16* hn       = (s & 1) ? hA : hB;
        cell_step_kernel<<<dim3(HH / 16, BB / 128), blk, 0, stream>>>(
            x_bf, wi2h_bf, b_i2h, wh2h_bf, b_h2h, att, hp, hn, s);
    }

    // ---- final classifier (s=127 odd -> last write went to hA) ----
    fc_kernel<<<dim3(CC / 64, BB / 128), blk, 0, stream>>>(
        hA, wfc_bf, b_fc, (float*)d_out);
}
